// TreeAwareTransformerEncoderLayer_86818468921668
// MI455X (gfx1250) — matrix operand order repaired and checked by reading, not compiled, not run on hardware
//
#include <hip/hip_runtime.h>

// Problem constants (fixed by the reference)
#define B_   4
#define L_   1024
#define D_   1024
#define H_   16
#define HD_  64
#define DFF_ 4096

typedef __attribute__((ext_vector_type(16))) __bf16 v16bf;
typedef __attribute__((ext_vector_type(8)))  float  v8f;

union BF16Frag {
  v16bf v;
  unsigned short u[16];
  uint4 q[2];
};

// f32 -> bf16 round-to-nearest-even (bit pattern only; no __bf16 math)
__device__ __forceinline__ unsigned short f2bf(float f) {
  unsigned int u = __float_as_uint(f);
  unsigned int r = (u + 0x7FFFu + ((u >> 16) & 1u)) >> 16;
  return (unsigned short)r;
}
__device__ __forceinline__ unsigned int pack2bf(float a, float b) {
  return (unsigned int)f2bf(a) | ((unsigned int)f2bf(b) << 16);
}
__device__ __forceinline__ uint4 pack8bf(float4 a, float4 b) {
  uint4 r;
  r.x = pack2bf(a.x, a.y); r.y = pack2bf(a.z, a.w);
  r.z = pack2bf(b.x, b.y); r.w = pack2bf(b.z, b.w);
  return r;
}

// CDNA5 async global->LDS copy of 16 bytes (per lane), GVS addressing:
// mem = SADDR(u64) + VADDR(i32); LDS dest = per-lane LDS byte offset.
// Tracked by ASYNCcnt.
__device__ __forceinline__ void async_copy_b128(void* lds_ptr,
                                                unsigned long long gbase,
                                                unsigned byte_off) {
  unsigned lds_off = (unsigned)(unsigned long long)lds_ptr;
  asm volatile("global_load_async_to_lds_b128 %0, %1, %2"
               :: "v"(lds_off), "v"(byte_off), "s"(gbase)
               : "memory");
}
__device__ __forceinline__ void wait_async0() {
  asm volatile("s_wait_asynccnt 0x0" ::: "memory");
}

// ---------------------------------------------------------------------------
// Pre-pass: f32 -> bf16 (8 elements/thread, vectorized)
// ---------------------------------------------------------------------------
__global__ __launch_bounds__(256) void convert_bf16_kernel(
    const float* __restrict__ x, unsigned short* __restrict__ y) {
  int i = (blockIdx.x * 256 + threadIdx.x) * 8;
  const float4* x4 = (const float4*)(x + i);
  *(uint4*)(y + i) = pack8bf(x4[0], x4[1]);
}

// ---------------------------------------------------------------------------
// Pre-pass: W[K,N] f32 -> Wt[N,K] bf16 (tiled LDS transpose, coalesced)
// grid = (K/32, N/32), block = 256
// ---------------------------------------------------------------------------
__global__ __launch_bounds__(256) void transpose_convert_kernel(
    const float* __restrict__ W, unsigned short* __restrict__ Wt, int K, int N) {
  __shared__ unsigned short t[32][33];
  const int k0 = blockIdx.x * 32, n0 = blockIdx.y * 32;
  const int tx = threadIdx.x & 31, ty = threadIdx.x >> 5;  // ty: 0..7
#pragma unroll
  for (int r = 0; r < 32; r += 8)
    t[ty + r][tx] = f2bf(W[(size_t)(k0 + ty + r) * N + n0 + tx]);
  __syncthreads();
#pragma unroll
  for (int r = 0; r < 32; r += 8)
    Wt[(size_t)(n0 + ty + r) * K + k0 + tx] = t[tx][ty + r];
}

// ---------------------------------------------------------------------------
// GEMM: C = act(A @ W + bias) with A bf16 [M,K], Wt bf16 [N,K] (pre-transposed).
// Compile-time output mode (OUTF -> f32, OUTB -> bf16).
// Block = 256 thr = 8 waves (4m x 2n); block tile 128x128; wave tile 32x64
// (2x4 WMMA frags -> 8 WMMA per K-step, 12 ds_load_b128 = 1.5 loads/WMMA).
// BK=32, DOUBLE-BUFFERED: async global->LDS copies for tile i+1 issued before
// computing tile i, overlapping the DMA with the WMMAs.
// ---------------------------------------------------------------------------
template <int ACT, bool OUTF, bool OUTB>
__global__ __launch_bounds__(256) void gemm_bf16_kernel(
    const unsigned short* __restrict__ A, const unsigned short* __restrict__ Wt,
    const float* __restrict__ bias, float* __restrict__ Cf,
    unsigned short* __restrict__ Cb, int M, int N, int K) {
  __shared__ unsigned short As[2][128][40];      // [buf][m][k]
  __shared__ unsigned short Bs[2][8][16][40];    // [buf][n-tile][n][k]

  const int tid  = threadIdx.x;
  const int lane = tid & 31;
  const int wave = tid >> 5;
  const int hl   = lane >> 4;
  const int lq   = lane & 15;
  const int wm   = wave >> 1;    // 0..3
  const int wn   = wave & 1;     // 0..1
  const int bm   = blockIdx.x * 128;
  const int bn   = blockIdx.y * 128;

  // staging assignments (constant per thread)
  const int sa_m  = tid >> 1;           // 0..127
  const int sa_k  = (tid & 1) * 16;     // 0 or 16
  const int sb_n  = tid & 127;
  const int sb_kh = (tid >> 7) * 16;    // 0 or 16
  const int sb_t  = sb_n >> 4;
  const int sb_nn = sb_n & 15;

  const unsigned long long abase = (unsigned long long)A;
  const unsigned long long wbase = (unsigned long long)Wt;
  const unsigned a_off0 = ((unsigned)(bm + sa_m) * (unsigned)K + sa_k) * 2u;
  const unsigned w_off0 = ((unsigned)(bn + sb_n) * (unsigned)K + sb_kh) * 2u;

  auto stage = [&](int kc, int bufi) {
    unsigned ak = a_off0 + (unsigned)kc * 2u;
    async_copy_b128(&As[bufi][sa_m][sa_k], abase, ak);
    async_copy_b128(&As[bufi][sa_m][sa_k + 8], abase, ak + 16u);
    unsigned wk = w_off0 + (unsigned)kc * 2u;
    async_copy_b128(&Bs[bufi][sb_t][sb_nn][sb_kh], wbase, wk);
    async_copy_b128(&Bs[bufi][sb_t][sb_nn][sb_kh + 8], wbase, wk + 16u);
  };

  v8f acc[2][4] = {};
  stage(0, 0);
  int cur = 0;

  for (int kc = 0; kc < K; kc += 32) {
    wait_async0();       // this wave's copies into buf[cur] complete
    __syncthreads();     // all waves' copies visible; prev buf fully consumed
    if (kc + 32 < K) stage(kc + 32, cur ^ 1);  // overlap with compute below

    BF16Frag af[2], bf[4];
#pragma unroll
    for (int s = 0; s < 2; s++) {
      int m = wm * 32 + s * 16 + lq;
      af[s].q[0] = *(const uint4*)&As[cur][m][8 * hl];
      af[s].q[1] = *(const uint4*)&As[cur][m][16 + 8 * hl];
    }
#pragma unroll
    for (int s = 0; s < 4; s++) {
      int t = wn * 4 + s;
      bf[s].q[0] = *(const uint4*)&Bs[cur][t][lq][8 * hl];
      bf[s].q[1] = *(const uint4*)&Bs[cur][t][lq][16 + 8 * hl];
    }
#pragma unroll
    for (int sm = 0; sm < 2; sm++)
#pragma unroll
      for (int sn = 0; sn < 4; sn++)
        acc[sm][sn] = __builtin_amdgcn_wmma_f32_16x16x32_bf16(
            false, af[sm].v, false, bf[sn].v, (short)0, acc[sm][sn], false, false);
    cur ^= 1;
  }

  // epilogue: pure 32-bit address math, no runtime output branches
#pragma unroll
  for (int sm = 0; sm < 2; sm++)
#pragma unroll
    for (int sn = 0; sn < 4; sn++) {
      unsigned n = (unsigned)(bn + wn * 64 + sn * 16 + lq);
      float bv = bias[n];
      unsigned off = (unsigned)(bm + wm * 32 + sm * 16 + hl * 8) * (unsigned)N + n;
#pragma unroll
      for (int r = 0; r < 8; r++) {
        float vv = acc[sm][sn][r] + bv;
        if (ACT == 1) vv = fmaxf(vv, 0.f);
        if (OUTF) Cf[off] = vv;
        if (OUTB) Cb[off] = f2bf(vv);
        off += (unsigned)N;
      }
    }
}

// ---------------------------------------------------------------------------
// Flash attention over one head, bf16 in / bf16 out. q/k/v/ctx are [B, L, D]
// with head h in columns [h*64, h*64+64). K tile async-staged, V tile
// transposed into LDS; both DOUBLE-BUFFERED and shared by all 8 waves.
// Each wave owns 16 query rows with online softmax; 1/8 applied to scores.
// grid = (L/128, B*H), block = 256.
// ---------------------------------------------------------------------------
template <bool MASKED>
__global__ __launch_bounds__(256) void flash_attn_kernel(
    const unsigned short* __restrict__ q, const unsigned short* __restrict__ k,
    const unsigned short* __restrict__ v, const unsigned char* __restrict__ pmask,
    const unsigned char* __restrict__ smask, unsigned short* __restrict__ ctx) {
  __shared__ unsigned short Ks[2][32][80];       // [buf][key][d]
  __shared__ unsigned short Vs[2][64][40];       // [buf][d][key] transposed
  __shared__ unsigned short Plds[8][16][40];     // per-wave P tile (16 x 32)

  const int tid  = threadIdx.x;
  const int lane = tid & 31;
  const int wave = tid >> 5;
  const int hl   = lane >> 4;
  const int lq   = lane & 15;
  const int bh   = blockIdx.y;
  const int b    = bh / H_;
  const int h    = bh % H_;
  const int qbase = blockIdx.x * 128 + wave * 16;

  const int sk_key = tid >> 3;          // 0..31
  const int sk_d   = (tid & 7) * 8;     // 0..56
  const int sv_d   = tid & 63;          // 0..63
  const int sv_k   = (tid >> 6) * 8;    // 0,8,16,24

  const unsigned long long kbase = (unsigned long long)k;
  const unsigned k_off0 =
      ((unsigned)(b * L_ + sk_key) * (unsigned)D_ + h * HD_ + sk_d) * 2u;

  auto stage = [&](int kj, int bufi) {
    async_copy_b128(&Ks[bufi][sk_key][sk_d], kbase,
                    k_off0 + (unsigned)(kj * D_) * 2u);
    const unsigned short* vp =
        v + (unsigned)(b * L_ + kj + sv_k) * D_ + h * HD_ + sv_d;
    unsigned short t8[8];
#pragma unroll
    for (int kk = 0; kk < 8; kk++) t8[kk] = vp[(unsigned)kk * D_];
    uint4 st;
    st.x = (unsigned)t8[0] | ((unsigned)t8[1] << 16);
    st.y = (unsigned)t8[2] | ((unsigned)t8[3] << 16);
    st.z = (unsigned)t8[4] | ((unsigned)t8[5] << 16);
    st.w = (unsigned)t8[6] | ((unsigned)t8[7] << 16);
    *(uint4*)&Vs[bufi][sv_d][sv_k] = st;
  };

  // Q fragments: two 16x32 A-frags covering HD=64 (plain b128 loads of bf16)
  BF16Frag aq[2];
  {
    const unsigned short* qrow =
        q + (unsigned)(b * L_ + qbase + lq) * D_ + h * HD_;
#pragma unroll
    for (int c = 0; c < 2; c++) {
      aq[c].q[0] = *(const uint4*)(qrow + c * 32 + 8 * hl);
      aq[c].q[1] = *(const uint4*)(qrow + c * 32 + 16 + 8 * hl);
    }
  }

  float mrow[8], lrow[8];
#pragma unroll
  for (int r = 0; r < 8; r++) { mrow[r] = -1e30f; lrow[r] = 0.f; }
  v8f of[4] = {};

  stage(0, 0);
  int cur = 0;

  for (int kj = 0; kj < L_; kj += 32) {
    wait_async0();
    __syncthreads();
    if (kj + 32 < L_) stage(kj + 32, cur ^ 1);  // overlap with compute below

    // S(16x32) = Q(16x64) @ K^T(64x32)
    v8f sc[2] = {};
#pragma unroll
    for (int sub = 0; sub < 2; sub++) {
      int krow = sub * 16 + lq;
#pragma unroll
      for (int c = 0; c < 2; c++) {
        BF16Frag bk;
        bk.q[0] = *(const uint4*)&Ks[cur][krow][c * 32 + 8 * hl];
        bk.q[1] = *(const uint4*)&Ks[cur][krow][c * 32 + 16 + 8 * hl];
        sc[sub] = __builtin_amdgcn_wmma_f32_16x16x32_bf16(
            false, aq[c].v, false, bk.v, (short)0, sc[sub], false, false);
      }
    }

    if (MASKED) {
#pragma unroll
      for (int sub = 0; sub < 2; sub++) {
        int key = kj + sub * 16 + lq;
#pragma unroll
        for (int r = 0; r < 8; r++) {
          int qr = qbase + r + hl * 8;
          unsigned mi = (unsigned)(b * L_ + qr) * (unsigned)L_ + key;
          if (!(pmask[mi] | smask[mi])) sc[sub][r] = -8e30f;
        }
      }
    }

    // online softmax; row reductions across the 16 lanes of each half
    float p0[8], p1[8];
#pragma unroll
    for (int r = 0; r < 8; r++) {
      float s0 = sc[0][r] * 0.125f, s1 = sc[1][r] * 0.125f;
      float cm = fmaxf(s0, s1);
#pragma unroll
      for (int off = 8; off >= 1; off >>= 1) cm = fmaxf(cm, __shfl_xor(cm, off));
      float mn = fmaxf(mrow[r], cm);
      float e0 = __expf(s0 - mn), e1 = __expf(s1 - mn);
      float rs = e0 + e1;
#pragma unroll
      for (int off = 8; off >= 1; off >>= 1) rs += __shfl_xor(rs, off);
      float corr = __expf(mrow[r] - mn);
      lrow[r] = lrow[r] * corr + rs;
      mrow[r] = mn;
      p0[r] = e0; p1[r] = e1;
#pragma unroll
      for (int c = 0; c < 4; c++) of[c][r] *= corr;
    }

    // P: C-fragment layout -> A-fragment layout via per-wave LDS tile
#pragma unroll
    for (int r = 0; r < 8; r++) {
      int row = r + hl * 8;
      Plds[wave][row][lq]      = f2bf(p0[r]);
      Plds[wave][row][16 + lq] = f2bf(p1[r]);
    }
    BF16Frag pa;
    pa.q[0] = *(const uint4*)&Plds[wave][lq][8 * hl];
    pa.q[1] = *(const uint4*)&Plds[wave][lq][16 + 8 * hl];

    // O(16x64) += P(16x32) @ V(32x64)
#pragma unroll
    for (int c = 0; c < 4; c++) {
      BF16Frag bv;
      bv.q[0] = *(const uint4*)&Vs[cur][c * 16 + lq][8 * hl];
      bv.q[1] = *(const uint4*)&Vs[cur][c * 16 + lq][16 + 8 * hl];
      of[c] = __builtin_amdgcn_wmma_f32_16x16x32_bf16(
          false, pa.v, false, bv.v, (short)0, of[c], false, false);
    }
    cur ^= 1;
  }

  // epilogue: normalize and store bf16 (32-bit offsets)
#pragma unroll
  for (int r = 0; r < 8; r++) {
    unsigned m = (unsigned)(qbase + r + hl * 8);
    float inv = 1.f / lrow[r];
    unsigned off = ((unsigned)b * L_ + m) * (unsigned)D_ + h * HD_ + lq;
#pragma unroll
    for (int c = 0; c < 4; c++)
      ctx[off + c * 16] = f2bf(of[c][r] * inv);
  }
}

// ---------------------------------------------------------------------------
// out = LayerNorm(x + res) * g + b ; optional bf16 copy. One block per row.
// ---------------------------------------------------------------------------
template <bool OUTB>
__global__ __launch_bounds__(256) void add_ln_kernel(
    const float* __restrict__ x, const float* __restrict__ res,
    const float* __restrict__ g, const float* __restrict__ bta,
    float* __restrict__ out, unsigned short* __restrict__ outb) {
  __shared__ float sred[16];
  const int row = blockIdx.x;
  const int tid = threadIdx.x;
  const float4* xr = (const float4*)(x + (size_t)row * D_);
  const float4* rr = (const float4*)(res + (size_t)row * D_);
  const float4* g4 = (const float4*)g;
  const float4* b4 = (const float4*)bta;
  float4* o4 = (float4*)(out + (size_t)row * D_);

  float4 xv = xr[tid], rv = rr[tid];
  float4 vv;
  vv.x = xv.x + rv.x; vv.y = xv.y + rv.y;
  vv.z = xv.z + rv.z; vv.w = xv.w + rv.w;
  float s1 = vv.x + vv.y + vv.z + vv.w;
  float s2 = vv.x * vv.x + vv.y * vv.y + vv.z * vv.z + vv.w * vv.w;
#pragma unroll
  for (int off = 16; off >= 1; off >>= 1) {
    s1 += __shfl_xor(s1, off);
    s2 += __shfl_xor(s2, off);
  }
  int wid = tid >> 5, ln = tid & 31;
  if (ln == 0) { sred[wid] = s1; sred[8 + wid] = s2; }
  __syncthreads();
  if (tid == 0) {
    float a = 0.f, c2 = 0.f;
    for (int i = 0; i < 8; i++) { a += sred[i]; c2 += sred[8 + i]; }
    sred[0] = a; sred[8] = c2;
  }
  __syncthreads();
  float mean = sred[0] / (float)D_;
  float var  = sred[8] / (float)D_ - mean * mean;
  float rstd = rsqrtf(var + 1e-5f);
  float4 gv = g4[tid], bv = b4[tid], ov;
  ov.x = (vv.x - mean) * rstd * gv.x + bv.x;
  ov.y = (vv.y - mean) * rstd * gv.y + bv.y;
  ov.z = (vv.z - mean) * rstd * gv.z + bv.z;
  ov.w = (vv.w - mean) * rstd * gv.w + bv.w;
  o4[tid] = ov;
  if (OUTB) {
    uint2 p;
    p.x = pack2bf(ov.x, ov.y);
    p.y = pack2bf(ov.z, ov.w);
    ((uint2*)(outb + (size_t)row * D_))[tid] = p;
  }
}

// ---------------------------------------------------------------------------
extern "C" void kernel_launch(void* const* d_in, const int* in_sizes, int n_in,
                              void* d_out, int out_size, void* d_ws, size_t ws_size,
                              hipStream_t stream) {
  (void)in_sizes; (void)n_in; (void)out_size; (void)ws_size;

  const float* src   = (const float*)d_in[0];
  const unsigned char* pmask = (const unsigned char*)d_in[1];
  const unsigned char* smask = (const unsigned char*)d_in[2];
  const float* wf[10] = {
      (const float*)d_in[3],  (const float*)d_in[5],  (const float*)d_in[7],
      (const float*)d_in[9],  (const float*)d_in[11], (const float*)d_in[13],
      (const float*)d_in[15], (const float*)d_in[17], (const float*)d_in[19],
      (const float*)d_in[21]};
  const float* mha_bq = (const float*)d_in[4];
  const float* mha_bk = (const float*)d_in[6];
  const float* mha_bv = (const float*)d_in[8];
  const float* mha_bo = (const float*)d_in[10];
  const float* tre_bq = (const float*)d_in[12];
  const float* tre_bk = (const float*)d_in[14];
  const float* tre_bv = (const float*)d_in[16];
  const float* tre_bo = (const float*)d_in[18];
  const float* b1 = (const float*)d_in[20];
  const float* b2 = (const float*)d_in[22];
  const float* ln1_g = (const float*)d_in[23], *ln1_b = (const float*)d_in[24];
  const float* ln2_g = (const float*)d_in[25], *ln2_b = (const float*)d_in[26];
  const float* ln3_g = (const float*)d_in[27], *ln3_b = (const float*)d_in[28];

  // workspace carve-up
  char* wsp = (char*)d_ws;
  const size_t SZA = (size_t)B_ * L_ * D_;          // 4M elements
  auto take = [&](size_t bytes) {
    char* p = wsp;
    wsp += (bytes + 255) & ~(size_t)255;
    return p;
  };
  unsigned short* srcb = (unsigned short*)take(SZA * 2);
  unsigned short* wt[10];
  for (int i = 0; i < 8; i++) wt[i] = (unsigned short*)take((size_t)D_ * D_ * 2);
  wt[8] = (unsigned short*)take((size_t)D_ * DFF_ * 2);   // w1t [DFF][D]
  wt[9] = (unsigned short*)take((size_t)DFF_ * D_ * 2);   // w2t [D][DFF]
  unsigned short* qb   = (unsigned short*)take(SZA * 2);
  unsigned short* kb   = (unsigned short*)take(SZA * 2);
  unsigned short* vb   = (unsigned short*)take(SZA * 2);
  unsigned short* ctxb = (unsigned short*)take(SZA * 2);
  float*          x1   = (float*)take(SZA * 4);
  unsigned short* x1b  = (unsigned short*)take(SZA * 2);
  float*          x2   = (float*)take(SZA * 4);
  unsigned short* x2b  = (unsigned short*)take(SZA * 2);
  float*          tmp  = (float*)take(SZA * 4);
  unsigned short* ffhb = qb;  // 32MB, reuses qb/kb/vb/ctxb after last use

  const int M = B_ * L_;                      // 4096
  dim3 blk(256);
  dim3 gproj(M / 128, D_ / 128);              // (32, 8)
  dim3 gattn(L_ / 128, B_ * H_);              // (8, 64)
  dim3 gff1(M / 128, DFF_ / 128);             // (32, 32)

  // ---- pre-pass: bf16 conversions / weight transposes ----
  convert_bf16_kernel<<<dim3(SZA / 2048), blk, 0, stream>>>(src, srcb);
  for (int i = 0; i < 8; i++)
    transpose_convert_kernel<<<dim3(D_ / 32, D_ / 32), blk, 0, stream>>>(
        wf[i], wt[i], D_, D_);
  transpose_convert_kernel<<<dim3(D_ / 32, DFF_ / 32), blk, 0, stream>>>(
      wf[8], wt[8], D_, DFF_);
  transpose_convert_kernel<<<dim3(DFF_ / 32, D_ / 32), blk, 0, stream>>>(
      wf[9], wt[9], DFF_, D_);

  // ---- standard MHA ----
  gemm_bf16_kernel<0, false, true><<<gproj, blk, 0, stream>>>(srcb, wt[0], mha_bq, nullptr, qb, M, D_, D_);
  gemm_bf16_kernel<0, false, true><<<gproj, blk, 0, stream>>>(srcb, wt[1], mha_bk, nullptr, kb, M, D_, D_);
  gemm_bf16_kernel<0, false, true><<<gproj, blk, 0, stream>>>(srcb, wt[2], mha_bv, nullptr, vb, M, D_, D_);
  flash_attn_kernel<false><<<gattn, blk, 0, stream>>>(qb, kb, vb, nullptr, nullptr, ctxb);
  gemm_bf16_kernel<0, true, false><<<gproj, blk, 0, stream>>>(ctxb, wt[3], mha_bo, tmp, nullptr, M, D_, D_);
  add_ln_kernel<true><<<dim3(M), blk, 0, stream>>>(src, tmp, ln1_g, ln1_b, x1, x1b);

  // ---- tree-guided MHA (mask = parent | sibling) ----
  gemm_bf16_kernel<0, false, true><<<gproj, blk, 0, stream>>>(x1b, wt[4], tre_bq, nullptr, qb, M, D_, D_);
  gemm_bf16_kernel<0, false, true><<<gproj, blk, 0, stream>>>(x1b, wt[5], tre_bk, nullptr, kb, M, D_, D_);
  gemm_bf16_kernel<0, false, true><<<gproj, blk, 0, stream>>>(x1b, wt[6], tre_bv, nullptr, vb, M, D_, D_);
  flash_attn_kernel<true><<<gattn, blk, 0, stream>>>(qb, kb, vb, pmask, smask, ctxb);
  gemm_bf16_kernel<0, true, false><<<gproj, blk, 0, stream>>>(ctxb, wt[7], tre_bo, tmp, nullptr, M, D_, D_);
  add_ln_kernel<true><<<dim3(M), blk, 0, stream>>>(x1, tmp, ln2_g, ln2_b, x2, x2b);

  // ---- FFN ----
  gemm_bf16_kernel<1, false, true><<<gff1, blk, 0, stream>>>(x2b, wt[8], b1, nullptr, ffhb, M, DFF_, D_);
  gemm_bf16_kernel<0, true, false><<<gproj, blk, 0, stream>>>(ffhb, wt[9], b2, tmp, nullptr, M, D_, DFF_);
  add_ln_kernel<false><<<dim3(M), blk, 0, stream>>>(x2, tmp, ln3_g, ln3_b, (float*)d_out, nullptr);
}
